// MultiQueryAttention_20169166422801
// MI455X (gfx1250) — matrix operand order repaired and checked by reading, not compiled, not run on hardware
//
#include <hip/hip_runtime.h>
#include <hip/hip_bf16.h>

// ---------------------------------------------------------------------------
// MQA on gfx1250: bf16 WMMA everywhere, flash-fused attention,
// async global->LDS staging where the copy is layout-preserving.
//   x:[B,S,D] f32 ; wq:[D,D] bq:[D] ; wk/wv:[D,64] bk/bv:[64] ; wo:[D,D] bo:[D]
//   B=2 S=2048 D=1024 H=16 Hd=64
// ---------------------------------------------------------------------------

typedef __attribute__((ext_vector_type(16))) __bf16 v16bf;
typedef __attribute__((ext_vector_type(8)))  __bf16 v8bf;
typedef __attribute__((ext_vector_type(8)))  float  v8f;
typedef __attribute__((ext_vector_type(4)))  int    v4i;

#define CAT16(lo, hi) __builtin_shufflevector((lo), (hi), 0,1,2,3,4,5,6,7,8,9,10,11,12,13,14,15)

static __device__ __forceinline__ v8f wmma_bf16(v16bf a, v16bf b, v8f c) {
    // D = A(16x32 bf16) * B(32x16 bf16) + C(16x16 f32)
    return __builtin_amdgcn_wmma_f32_16x16x32_bf16(
        /*neg_a=*/false, a, /*neg_b=*/false, b,
        /*c_mod=*/(short)0, c, /*reuse_a=*/false, /*reuse_b=*/false);
}

// ---- async global->LDS copy (CDNA5 GLOBAL_LOAD_ASYNC_TO_LDS_B128) ---------
static __device__ __forceinline__ void async_copy_b128(const __bf16* g, __bf16* l) {
#if __has_builtin(__builtin_amdgcn_global_load_async_to_lds_b128)
    // signature (from compiler diagnostic): (v4i AS1*, v4i AS3*, imm offset, imm cpol)
    __attribute__((address_space(1))) v4i* gp =
        (__attribute__((address_space(1))) v4i*)(void*)g;
    __attribute__((address_space(3))) v4i* lp =
        (__attribute__((address_space(3))) v4i*)(void*)l;
    __builtin_amdgcn_global_load_async_to_lds_b128(gp, lp, /*offset=*/0, /*cpol=*/0);
#else
    *(v8bf*)l = *(const v8bf*)g;   // fallback: VGPR round trip
#endif
}

static __device__ __forceinline__ void wait_async0() {
#if __has_builtin(__builtin_amdgcn_global_load_async_to_lds_b128)
#if __has_builtin(__builtin_amdgcn_s_wait_asynccnt)
    __builtin_amdgcn_s_wait_asynccnt(0);
#else
    asm volatile("s_wait_asynccnt 0x0" ::: "memory");
#endif
#endif
}

// ----------------------------------------------------------------- constants
constexpr int kB  = 2;
constexpr int kS  = 2048;
constexpr int kD  = 1024;
constexpr int kH  = 16;
constexpr int kHd = 64;
constexpr int kM  = kB * kS;          // 4096 rows in all GEMMs

// ------------------------------------------------------------- f32 -> bf16
__global__ void cvt_f32_bf16(const float* __restrict__ src,
                             __bf16* __restrict__ dst, int n) {
    int i = blockIdx.x * blockDim.x + threadIdx.x;
    int stride = gridDim.x * blockDim.x;
    for (; i < n; i += stride) dst[i] = (__bf16)src[i];
}

// ---------------------------------------------------------------------------
// Generic GEMM: C[M,N] = (A[M,K] @ B[K,N] + bias[N]) * scale
//   A,B row-major bf16.  Block tile 128 x BN, 256 threads = 8 waves,
//   wave w computes rows [w*16, w*16+16) x BN cols (BN/16 C fragments).
//   B block staged transposed in LDS so B fragments are contiguous reads.
// ---------------------------------------------------------------------------
template <int BN, bool OUT_BF16>
__global__ __launch_bounds__(256)
void gemm_bf16_wmma(const __bf16* __restrict__ A, const __bf16* __restrict__ Bm,
                    const float* __restrict__ bias, void* __restrict__ Cout,
                    int M, int N, int K, float scale) {
    constexpr int NT = BN / 16;                   // C fragments per wave
    alignas(16) __shared__ __bf16 sBT[BN * 32];   // BT[n][k], k contiguous

    const int tid   = threadIdx.x;
    const int wave  = tid >> 5;
    const int lane  = tid & 31;
    const int lhalf = lane >> 4;      // 0: lanes 0-15, 1: lanes 16-31
    const int l16   = lane & 15;
    const int row0  = blockIdx.y * 128;
    const int col0  = blockIdx.x * BN;
    const int arow  = row0 + wave * 16 + l16;     // row this lane's A frag covers

    v8f acc[NT] = {};

    for (int kc = 0; kc < K; kc += 32) {
        // ---- stage B[kc:kc+32, col0:col0+BN] transposed into LDS ----------
        {
            const int k  = tid >> 3;               // 0..31
            const int n0 = (tid & 7) * (BN / 8);   // BN/8 cols per thread
#pragma unroll
            for (int c = 0; c < BN / 8; c += 8) {
                v8bf bv = *(const v8bf*)(Bm + (size_t)(kc + k) * N + (col0 + n0 + c));
#pragma unroll
                for (int j = 0; j < 8; ++j) sBT[(n0 + c + j) * 32 + k] = bv[j];
            }
            if (kc + 32 < K)
                __builtin_prefetch(Bm + (size_t)(kc + 32 + k) * N + (col0 + n0), 0, 1);
        }
        __syncthreads();

        // ---- A fragment (16x32), straight from global ---------------------
        // lanes 0-15 hold K {0..7,16..23}; lanes 16-31 hold K {8..15,24..31}
        const __bf16* gA = A + (size_t)arow * K + kc + (lhalf ? 8 : 0);
        v8bf a_lo = *(const v8bf*)gA;
        v8bf a_hi = *(const v8bf*)(gA + 16);
        v16bf afrag = CAT16(a_lo, a_hi);

#pragma unroll
        for (int nt = 0; nt < NT; ++nt) {
            // B fragment: lane column n = nt*16+l16, k {0..7,16..23} / {8..15,24..31} per lane half
            const __bf16* lB = &sBT[(nt * 16 + l16) * 32 + (lhalf ? 8 : 0)];
            v8bf b_lo = *(const v8bf*)lB;
            v8bf b_hi = *(const v8bf*)(lB + 16);
            acc[nt] = wmma_bf16(afrag, CAT16(b_lo, b_hi), acc[nt]);
        }
        __syncthreads();
    }

    // ---- epilogue: C frag lane = N, vgpr r = row (r, r+8 per lane half) ---
#pragma unroll
    for (int nt = 0; nt < NT; ++nt) {
        const int n  = col0 + nt * 16 + l16;
        const float bv = bias[n];
#pragma unroll
        for (int r = 0; r < 8; ++r) {
            const int m = row0 + wave * 16 + r + lhalf * 8;
            const float val = (acc[nt][r] + bv) * scale;
            if (OUT_BF16) ((__bf16*)Cout)[(size_t)m * N + n] = (__bf16)val;
            else          ((float*) Cout)[(size_t)m * N + n] = val;
        }
    }
}

// ---------------------------------------------------------------------------
// Flash MQA core.  Q:[B*S,D] bf16 (already scaled by 1/sqrt(D) and biased),
// Kb,Vb:[B*S,Hd] bf16 shared across heads.  O:[B*S,D] bf16 = softmax(QK^T)V.
// Block: 256 thr = 8 waves, each wave owns a 16-query tile (128 q / block).
// Key loop: 32 keys / iteration; K tile staged via async global->LDS DMA,
// V tile staged transposed through VGPRs; both shared by all 8 waves.
// ---------------------------------------------------------------------------
__global__ __launch_bounds__(256)
void mqa_flash(const __bf16* __restrict__ Q, const __bf16* __restrict__ Kb,
               const __bf16* __restrict__ Vb, __bf16* __restrict__ O) {
    alignas(16) __shared__ __bf16 sK[32 * 64];        // row-major [key][d]
    alignas(16) __shared__ __bf16 sVT[64 * 32];       // transposed [d][key]
    alignas(16) __shared__ __bf16 sP[8 * 16 * 32];    // per-wave P tile [q][key]

    const int tid   = threadIdx.x;
    const int wave  = tid >> 5;
    const int lane  = tid & 31;
    const int lhalf = lane >> 4;
    const int l16   = lane & 15;
    const int b     = blockIdx.z;
    const int h     = blockIdx.y;
    const int q0    = blockIdx.x * 128 + wave * 16;

    // ---- load Q fragments for this wave's 16 rows (Hd=64 -> 2 K-chunks) ---
    v16bf qf[2];
    {
        const size_t qbase = ((size_t)(b * kS + q0 + l16)) * kD + h * kHd;
#pragma unroll
        for (int f = 0; f < 2; ++f) {
            const __bf16* g = Q + qbase + f * 32 + (lhalf ? 8 : 0);
            v8bf lo = *(const v8bf*)g;
            v8bf hi = *(const v8bf*)(g + 16);
            qf[f] = CAT16(lo, hi);
        }
    }

    v8f acc[4] = {};                    // 16(q) x 64(d) output accumulator
    float rmax[8], rsum[8];
#pragma unroll
    for (int r = 0; r < 8; ++r) { rmax[r] = -3.0e38f; rsum[r] = 0.0f; }

    const __bf16* kbase = Kb + (size_t)b * kS * kHd;
    const __bf16* vbase = Vb + (size_t)b * kS * kHd;
    __bf16* myP = &sP[wave * 16 * 32];

    for (int kb0 = 0; kb0 < kS; kb0 += 32) {
        // ---- cooperative stage: K async DMA, V transposed through VGPRs ---
        {
            const int kr = tid >> 3;          // 0..31 key row
            const int d0 = (tid & 7) * 8;     // 0..56
            async_copy_b128(kbase + (size_t)(kb0 + kr) * kHd + d0,
                            &sK[kr * 64 + d0]);
            v8bf vv = *(const v8bf*)(vbase + (size_t)(kb0 + kr) * kHd + d0);
#pragma unroll
            for (int j = 0; j < 8; ++j) sVT[(d0 + j) * 32 + kr] = vv[j];
            if (kb0 + 32 < kS) {
                __builtin_prefetch(kbase + (size_t)(kb0 + 32 + kr) * kHd + d0, 0, 1);
                __builtin_prefetch(vbase + (size_t)(kb0 + 32 + kr) * kHd + d0, 0, 1);
            }
        }
        wait_async0();                    // our async chunk landed in LDS
        __syncthreads();                  // everyone's chunks visible

        // ---- scores: two 16-key subtiles, Hd=64 contraction (2 wmma each) -
        v8f sc[2] = {};
#pragma unroll
        for (int kt = 0; kt < 2; ++kt) {
#pragma unroll
            for (int f = 0; f < 2; ++f) {
                // B frag: lane column = key kt*16+l16; d f*32+{0..7,16..23} / +{8..15,24..31} per lane half
                const __bf16* lb = &sK[(kt * 16 + l16) * 64 + f * 32 + (lhalf ? 8 : 0)];
                v8bf lo = *(const v8bf*)lb;
                v8bf hi = *(const v8bf*)(lb + 16);
                sc[kt] = wmma_bf16(qf[f], CAT16(lo, hi), sc[kt]);
            }
        }

        // ---- online softmax (rows r / r+8 per lane half) ------------------
#pragma unroll
        for (int r = 0; r < 8; ++r) {
            float m = fmaxf(sc[0][r], sc[1][r]);
            m = fmaxf(m, __shfl_xor(m, 8, 16));
            m = fmaxf(m, __shfl_xor(m, 4, 16));
            m = fmaxf(m, __shfl_xor(m, 2, 16));
            m = fmaxf(m, __shfl_xor(m, 1, 16));
            const float newm = fmaxf(rmax[r], m);
            const float corr = __expf(rmax[r] - newm);
            const float p0 = __expf(sc[0][r] - newm);
            const float p1 = __expf(sc[1][r] - newm);
            float ps = p0 + p1;
            ps += __shfl_xor(ps, 8, 16);
            ps += __shfl_xor(ps, 4, 16);
            ps += __shfl_xor(ps, 2, 16);
            ps += __shfl_xor(ps, 1, 16);
            rsum[r] = rsum[r] * corr + ps;
            rmax[r] = newm;
#pragma unroll
            for (int dt = 0; dt < 4; ++dt) acc[dt][r] *= corr;
            // spill P (C-fragment layout) to this wave's LDS tile as bf16
            myP[(r + lhalf * 8) * 32 + l16]      = (__bf16)p0;
            myP[(r + lhalf * 8) * 32 + 16 + l16] = (__bf16)p1;
        }
        // wave-private LDS round-trip: make the 16 ds stores land first
        asm volatile("s_wait_dscnt 0x0" ::: "memory");

        // ---- reload P as an A fragment (16 q x 32 key) --------------------
        const __bf16* lp = &myP[l16 * 32 + (lhalf ? 8 : 0)];
        v8bf plo = *(const v8bf*)lp;
        v8bf phi = *(const v8bf*)(lp + 16);
        v16bf pfrag = CAT16(plo, phi);

        // ---- acc += P @ V  (4 d-tiles of 16) ------------------------------
#pragma unroll
        for (int dt = 0; dt < 4; ++dt) {
            const __bf16* lv = &sVT[(dt * 16 + l16) * 32 + (lhalf ? 8 : 0)];
            v8bf lo = *(const v8bf*)lv;
            v8bf hi = *(const v8bf*)(lv + 16);
            acc[dt] = wmma_bf16(pfrag, CAT16(lo, hi), acc[dt]);
        }
        __syncthreads();
    }

    // ---- normalize and write O (bf16, feeds the final projection) ---------
#pragma unroll
    for (int r = 0; r < 8; ++r) {
        const float inv = 1.0f / rsum[r];
        const int m = q0 + r + lhalf * 8;
#pragma unroll
        for (int dt = 0; dt < 4; ++dt) {
            O[((size_t)(b * kS + m)) * kD + h * kHd + dt * 16 + l16] =
                (__bf16)(acc[dt][r] * inv);
        }
    }
}

// ---------------------------------------------------------------------------
extern "C" void kernel_launch(void* const* d_in, const int* in_sizes, int n_in,
                              void* d_out, int out_size, void* d_ws, size_t ws_size,
                              hipStream_t stream) {
    (void)in_sizes; (void)n_in; (void)out_size; (void)ws_size;

    const float* x  = (const float*)d_in[0];
    const float* wq = (const float*)d_in[1];
    const float* bq = (const float*)d_in[2];
    const float* wk = (const float*)d_in[3];
    const float* bk = (const float*)d_in[4];
    const float* wv = (const float*)d_in[5];
    const float* bv = (const float*)d_in[6];
    const float* wo = (const float*)d_in[7];
    const float* bo = (const float*)d_in[8];
    float* out = (float*)d_out;

    // workspace layout (bf16 buffers), ~31 MB total -> L2 resident
    char* ws = (char*)d_ws;
    constexpr size_t oXB  = 0;                                   // x     4096x1024
    constexpr size_t oQB  = oXB  + (size_t)kM * kD * 2;          // q     4096x1024
    constexpr size_t oOB  = oQB  + (size_t)kM * kD * 2;          // attn  4096x1024
    constexpr size_t oWQB = oOB  + (size_t)kM * kD * 2;          // wq    1024x1024
    constexpr size_t oWOB = oWQB + (size_t)kD * kD * 2;          // wo    1024x1024
    constexpr size_t oKB  = oWOB + (size_t)kD * kD * 2;          // k     4096x64
    constexpr size_t oVB  = oKB  + (size_t)kM * kHd * 2;         // v     4096x64
    constexpr size_t oWKB = oVB  + (size_t)kM * kHd * 2;         // wk    1024x64
    constexpr size_t oWVB = oWKB + (size_t)kD * kHd * 2;         // wv    1024x64

    __bf16* xb  = (__bf16*)(ws + oXB);
    __bf16* qb  = (__bf16*)(ws + oQB);
    __bf16* ob  = (__bf16*)(ws + oOB);
    __bf16* wqb = (__bf16*)(ws + oWQB);
    __bf16* wob = (__bf16*)(ws + oWOB);
    __bf16* kb  = (__bf16*)(ws + oKB);
    __bf16* vb  = (__bf16*)(ws + oVB);
    __bf16* wkb = (__bf16*)(ws + oWKB);
    __bf16* wvb = (__bf16*)(ws + oWVB);

    // f32 -> bf16 conversions
    cvt_f32_bf16<<<512, 256, 0, stream>>>(x,  xb,  kM * kD);
    cvt_f32_bf16<<<256, 256, 0, stream>>>(wq, wqb, kD * kD);
    cvt_f32_bf16<<<64,  256, 0, stream>>>(wk, wkb, kD * kHd);
    cvt_f32_bf16<<<64,  256, 0, stream>>>(wv, wvb, kD * kHd);
    cvt_f32_bf16<<<256, 256, 0, stream>>>(wo, wob, kD * kD);

    // projections (1/sqrt(D)=1/32 folded into Q); wide tiles for D-wide GEMMs
    gemm_bf16_wmma<128, true><<<dim3(kD / 128, kM / 128), 256, 0, stream>>>(
        xb, wqb, bq, qb, kM, kD, kD, 1.0f / 32.0f);
    gemm_bf16_wmma<64, true><<<dim3(kHd / 64, kM / 128), 256, 0, stream>>>(
        xb, wkb, bk, kb, kM, kHd, kD, 1.0f);
    gemm_bf16_wmma<64, true><<<dim3(kHd / 64, kM / 128), 256, 0, stream>>>(
        xb, wvb, bv, vb, kM, kHd, kD, 1.0f);

    // flash attention: grid (q-tiles, heads, batch)
    mqa_flash<<<dim3(kS / 128, kH, kB), 256, 0, stream>>>(qb, kb, vb, ob);

    // output projection -> f32 d_out
    gemm_bf16_wmma<128, false><<<dim3(kD / 128, kM / 128), 256, 0, stream>>>(
        ob, wob, bo, out, kM, kD, kD, 1.0f);
}
